// BiLstmCrf_69595650064842
// MI455X (gfx1250) — compile-verified, matrix operand order checked by reading, not run on hardware
//
#include <hip/hip_runtime.h>
#include <hip/hip_bf16.h>

typedef __bf16 bf16;
typedef __attribute__((ext_vector_type(16))) __bf16 bf16x16;
typedef __attribute__((ext_vector_type(8)))  __bf16 bf16x8;
typedef __attribute__((ext_vector_type(8)))  float  floatx8;

#define B_  32
#define S_  512
#define D_  256
#define H_  256
#define T_  10
#define ROWS (B_ * S_)        // 16384, row = s*32 + b (time-major)

// ---------------- workspace layout (bytes) ----------------
#define OFF_APACK   0u                                   // 16384x256 bf16   = 8388608
#define OFF_WIHF    (OFF_APACK + 8388608u)               // 256x1024 bf16    = 524288
#define OFF_WHHF    (OFF_WIHF  + 524288u)
#define OFF_WIHB    (OFF_WHHF  + 524288u)
#define OFF_WHHB    (OFF_WIHB  + 524288u)
#define OFF_WLT     (OFF_WHHB  + 524288u)                // 512x16 bf16      = 16384
#define OFF_GF      (OFF_WLT   + 16384u)                 // 16384x1024 bf16  = 33554432
#define OFF_GB      (OFF_GF    + 33554432u)              // 16384x1024 bf16  = 33554432
#define OFF_HOUT    (OFF_GB    + 33554432u)              // 16384x512 f32    = 33554432
// overlaid on GF/GB after both LSTM passes complete:
#define OFF_HPACK   OFF_GF                               // 16384x512 bf16   = 16777216
#define OFF_LOGITS  (OFF_GF + 16777216u)                 // 16384x16 f32     = 1048576
#define OFF_HIST    (OFF_GF + 16777216u + 1048576u)      // 511x32x10 int

// ---- fast activations: gfx1250 has v_tanh_f32 (TRANS op)
#if __has_builtin(__builtin_amdgcn_tanhf)
__device__ __forceinline__ float fast_tanh(float x) { return __builtin_amdgcn_tanhf(x); }
#else
__device__ __forceinline__ float fast_tanh(float x) { return tanhf(x); }
#endif
__device__ __forceinline__ float fast_sigmoid(float x) { return 0.5f * fast_tanh(0.5f * x) + 0.5f; }

// K-index map inside a 16x32 bf16 A/B fragment: element e of lane l
// k_local = ((e>>3)<<4) + ((l>>4)<<3) + (e&7)

// ---- pack W[N][K] -> B fragments of W^T (K x Npad), frag = kt*(Npad/16)+nt
__global__ void pack_weightT(const float* __restrict__ W, bf16* __restrict__ out,
                             int K, int N, int Npad) {
    int NT = Npad >> 4;
    int total = (K >> 5) * NT * 32;
    int t = blockIdx.x * blockDim.x + threadIdx.x;
    if (t >= total) return;
    int lane = t & 31, frag = t >> 5;
    int nt = frag % NT, kt = frag / NT;
    int n = (nt << 4) + (lane & 15);
    int khalf = (lane >> 4) << 3;
    union { bf16x16 v; bf16 h[16]; } f;
#pragma unroll
    for (int e = 0; e < 16; ++e) {
        int k = (kt << 5) + ((e >> 3) << 4) + khalf + (e & 7);
        float v = (n < N) ? W[(size_t)n * K + k] : 0.0f;
        f.h[e] = (bf16)v;
    }
    *(bf16x16*)(out + ((size_t)frag * 32 + lane) * 16) = f.v;
}

// ---- embedding gather + pack into A fragments; row = s*32 + b; frag = mt*8 + kt
__global__ void pack_embed(const int* __restrict__ sents, const float* __restrict__ emb,
                           bf16* __restrict__ out) {
    int t = blockIdx.x * blockDim.x + threadIdx.x;     // (ROWS/16)*8*32 = 262144
    if (t >= (ROWS / 16) * 8 * 32) return;
    int lane = t & 31, frag = t >> 5;
    int kt = frag & 7, mt = frag >> 3;
    int row = (mt << 4) + (lane & 15);
    int s = row >> 5, b = row & 31;
    int tok = sents[b * S_ + s];
    const float* e = emb + (size_t)tok * D_;
    int khalf = (lane >> 4) << 3;
    union { bf16x16 v; bf16 h[16]; } f;
#pragma unroll
    for (int q = 0; q < 16; ++q) {
        int k = (kt << 5) + ((q >> 3) << 4) + khalf + (q & 7);
        f.h[q] = (bf16)e[k];
    }
    *(bf16x16*)(out + ((size_t)frag * 32 + lane) * 16) = f.v;
}

// ---- input-projection GEMM, BOTH directions (blockIdx.y = dir):
// G = A(16384x256) * WihT(256x1024) + bih + bhh, bf16 C-fragment output.
__global__ void __launch_bounds__(128) gemm_proj2(const bf16* __restrict__ A,
                                                  const bf16* __restrict__ BwF,
                                                  const bf16* __restrict__ BwB,
                                                  const float* __restrict__ bihF,
                                                  const float* __restrict__ bhhF,
                                                  const float* __restrict__ bihB,
                                                  const float* __restrict__ bhhB,
                                                  bf16* __restrict__ GoutF,
                                                  bf16* __restrict__ GoutB) {
    int dir = blockIdx.y;
    const bf16* Bw = dir ? BwB : BwF;
    const float* bih = dir ? bihB : bihF;
    const float* bhh = dir ? bhhB : bhhF;
    bf16* Gout = dir ? GoutB : GoutF;
    int lane = threadIdx.x & 31;
    int w = blockIdx.x * 4 + (threadIdx.x >> 5);   // 16384 waves per direction
    int mt = w >> 4;
    int nt0 = (w & 15) << 2;
    floatx8 acc[4];
#pragma unroll
    for (int j = 0; j < 4; ++j) {
        int n = ((nt0 + j) << 4) + (lane & 15);
        float bias = bih[n] + bhh[n];
#pragma unroll
        for (int r = 0; r < 8; ++r) acc[j][r] = bias;
    }
    bf16x16 aCur = *(const bf16x16*)(A + (((size_t)mt * 8 + 0) * 32 + lane) * 16);
    bf16x16 bCur[4];
#pragma unroll
    for (int j = 0; j < 4; ++j)
        bCur[j] = *(const bf16x16*)(Bw + (((size_t)0 * 64 + nt0 + j) * 32 + lane) * 16);
#pragma unroll 2
    for (int kt = 0; kt < 8; ++kt) {
        int kn = (kt < 7) ? (kt + 1) : 7;
        bf16x16 aNxt = *(const bf16x16*)(A + (((size_t)mt * 8 + kn) * 32 + lane) * 16);
        bf16x16 bNxt[4];
#pragma unroll
        for (int j = 0; j < 4; ++j)
            bNxt[j] = *(const bf16x16*)(Bw + (((size_t)kn * 64 + nt0 + j) * 32 + lane) * 16);
#pragma unroll
        for (int j = 0; j < 4; ++j)
            acc[j] = __builtin_amdgcn_wmma_f32_16x16x32_bf16(false, aCur, false, bCur[j],
                                                             (short)0, acc[j], false, false);
        aCur = aNxt;
#pragma unroll
        for (int j = 0; j < 4; ++j) bCur[j] = bNxt[j];
    }
#pragma unroll
    for (int j = 0; j < 4; ++j) {
        bf16x8 o;
#pragma unroll
        for (int r = 0; r < 8; ++r) o[r] = (bf16)acc[j][r];
        *(bf16x8*)(Gout + (((size_t)mt * 64 + nt0 + j) * 32 + lane) * 8) = o;
    }
}

// ---- recurrent BiLSTM: BOTH directions concurrently (blockIdx.x = dir),
// each block = 32 wave32 on one WGP, loops over time with LDS-resident state.
// Next timestep's G fragments are prefetched behind the activation phase.
__global__ void __launch_bounds__(1024) lstm_both(const bf16* __restrict__ Gf,
                                                  const bf16* __restrict__ Gb,
                                                  const bf16* __restrict__ WhhTf,
                                                  const bf16* __restrict__ WhhTb,
                                                  const int* __restrict__ lengths,
                                                  float* __restrict__ Hout) {
    int dir = blockIdx.x;
    const bf16* G   = dir ? Gb : Gf;
    const bf16* Whh = dir ? WhhTb : WhhTf;
    extern __shared__ char smem[];
    float* gates = (float*)smem;                        // 32*1024 f32 = 128K
    float* hS    = (float*)(smem + 131072);             // 32*256  f32 = 32K
    float* cS    = (float*)(smem + 131072 + 32768);     // 32*256  f32 = 32K
    bf16*  hbf   = (bf16*)(smem + 131072 + 65536);      // 32*256 bf16 = 16K
    int tid = threadIdx.x;
    int lane = tid & 31, w = tid >> 5;
    for (int i = tid; i < B_ * H_; i += 1024) { hS[i] = 0.f; cS[i] = 0.f; hbf[i] = (bf16)0.0f; }
    __syncthreads();
    // GEMM-phase roles: waves 0-15 -> mt 0 (batches 0-15), waves 16-31 -> mt 1
    int mtw = w >> 4;
    int nt0 = (w & 15) << 2;
    int m = (mtw << 4) + (lane & 15);
    int khalf = (lane >> 4) << 3;
    int nlo = lane & 15;
    int rofs = (lane >> 4) << 3;
    // activation-phase roles
    int bown = tid >> 5;
    int h0 = (tid & 31) * 8;
    int len = lengths[bown];
    // prime the G-fragment pipeline for the first timestep
    bf16x8 gNext[4];
    {
        int t0 = dir ? (S_ - 1) : 0;
#pragma unroll
        for (int j = 0; j < 4; ++j)
            gNext[j] = *(const bf16x8*)(G + ((((size_t)t0 * 2 + mtw) * 64 + nt0 + j) * 32 + lane) * 8);
    }
    for (int it = 0; it < S_; ++it) {
        int t = dir ? (S_ - 1 - it) : it;
        // ---- GEMM phase: gates = G[t] + h @ WhhT (one A fragment feeds 4 N-tiles)
        floatx8 acc[4];
#pragma unroll
        for (int j = 0; j < 4; ++j) {
#pragma unroll
            for (int r = 0; r < 8; ++r) acc[j][r] = (float)gNext[j][r];
        }
        for (int kt = 0; kt < 8; ++kt) {
            // issue VMEM B loads first so they overlap the LDS A-fragment build
            bf16x16 bfr[4];
#pragma unroll
            for (int j = 0; j < 4; ++j)
                bfr[j] = *(const bf16x16*)(Whh + (((size_t)(kt << 6) + nt0 + j) * 32 + lane) * 16);
            union { bf16x16 v; unsigned u[8]; } a;
            const unsigned* hrow = (const unsigned*)(hbf + m * H_ + (kt << 5) + khalf);
#pragma unroll
            for (int q = 0; q < 8; ++q)
                a.u[q] = hrow[((q >> 2) << 3) + (q & 3)];
#pragma unroll
            for (int j = 0; j < 4; ++j)
                acc[j] = __builtin_amdgcn_wmma_f32_16x16x32_bf16(false, a.v, false, bfr[j],
                                                                 (short)0, acc[j], false, false);
        }
#pragma unroll
        for (int j = 0; j < 4; ++j) {
            int nn = ((nt0 + j) << 4) + nlo;
#pragma unroll
            for (int r = 0; r < 8; ++r) {
                int bb = (mtw << 4) + r + rofs;
                gates[bb * 1024 + nn] = acc[j][r];
            }
        }
        // prefetch NEXT timestep's G fragments; latency hides behind
        // the barrier + activation phase (registers live across barriers)
        {
            int itn = (it + 1 < S_) ? (it + 1) : it;
            int tn = dir ? (S_ - 1 - itn) : itn;
#pragma unroll
            for (int j = 0; j < 4; ++j)
                gNext[j] = *(const bf16x8*)(G + ((((size_t)tn * 2 + mtw) * 64 + nt0 + j) * 32 + lane) * 8);
        }
        __syncthreads();
        // ---- activation phase: 8 cells per thread (v_tanh_f32-based)
        bool valid = (t < len);
#pragma unroll
        for (int q = 0; q < 8; ++q) {
            int h = h0 + q;
            float gi = gates[bown * 1024 + h];
            float gf = gates[bown * 1024 + 256 + h];
            float gg = gates[bown * 1024 + 512 + h];
            float go = gates[bown * 1024 + 768 + h];
            float i_ = fast_sigmoid(gi), f_ = fast_sigmoid(gf), o_ = fast_sigmoid(go);
            float g_ = fast_tanh(gg);
            float cprev = cS[bown * H_ + h];
            float hprev = hS[bown * H_ + h];
            float c2 = f_ * cprev + i_ * g_;
            float h2 = o_ * fast_tanh(c2);
            float cn = valid ? c2 : cprev;
            float hn = valid ? h2 : hprev;
            cS[bown * H_ + h] = cn;
            hS[bown * H_ + h] = hn;
            hbf[bown * H_ + h] = (bf16)hn;
            Hout[((size_t)t * 32 + bown) * 512 + dir * 256 + h] = valid ? h2 : 0.0f;
        }
        __syncthreads();
    }
}

// ---- pack Hout (16384x512 f32) into A fragments; frag = mt*32 + kt
__global__ void pack_hout(const float* __restrict__ Hout, bf16* __restrict__ out) {
    int t = blockIdx.x * blockDim.x + threadIdx.x;      // (ROWS/16)*32*32 = 1048576
    if (t >= (ROWS / 16) * 32 * 32) return;
    int lane = t & 31, frag = t >> 5;
    int kt = frag & 31, mt = frag >> 5;
    int row = (mt << 4) + (lane & 15);
    int khalf = (lane >> 4) << 3;
    union { bf16x16 v; bf16 h[16]; } f;
#pragma unroll
    for (int e = 0; e < 16; ++e) {
        int k = (kt << 5) + ((e >> 3) << 4) + khalf + (e & 7);
        f.h[e] = (bf16)Hout[(size_t)row * 512 + k];
    }
    *(bf16x16*)(out + ((size_t)frag * 32 + lane) * 16) = f.v;
}

// ---- logits GEMM: logits(16384 x 16pad) = H(16384x512) @ WlT(512x16pad) + bl
__global__ void __launch_bounds__(128) gemm_logits(const bf16* __restrict__ A,
                                                   const bf16* __restrict__ Bw,
                                                   const float* __restrict__ bl,
                                                   float* __restrict__ logits) {
    int lane = threadIdx.x & 31;
    int mt = blockIdx.x * 4 + (threadIdx.x >> 5);   // 1024 waves
    int n = lane & 15;
    float bias = (n < T_) ? bl[n] : 0.0f;
    floatx8 acc;
#pragma unroll
    for (int r = 0; r < 8; ++r) acc[r] = bias;
    for (int kt = 0; kt < 32; ++kt) {
        bf16x16 a = *(const bf16x16*)(A + (((size_t)mt * 32 + kt) * 32 + lane) * 16);
        bf16x16 b = *(const bf16x16*)(Bw + ((size_t)kt * 32 + lane) * 16);
        acc = __builtin_amdgcn_wmma_f32_16x16x32_bf16(false, a, false, b,
                                                      (short)0, acc, false, false);
    }
#pragma unroll
    for (int r = 0; r < 8; ++r) {
        int row = (mt << 4) + r + ((lane >> 4) << 3);
        logits[(size_t)row * 16 + n] = acc[r];
    }
}

// ---- CRF negative log-likelihood body (one 320-thread block)
__device__ void crf_loss_body(const float* __restrict__ logits,
                              const int* __restrict__ labels,
                              const float* __restrict__ start_t,
                              const float* __restrict__ end_t,
                              const float* __restrict__ trans,
                              float* __restrict__ out_loss) {
    __shared__ float alpha[B_][T_], nxtb[B_][T_];
    __shared__ float tr[T_][T_], st[T_], en[T_];
    __shared__ float llh[B_];
    int tid = threadIdx.x;
    int b = tid / T_, j = tid % T_;
    if (tid < T_ * T_) tr[tid / T_][tid % T_] = trans[tid];
    if (tid < T_) { st[tid] = start_t[tid]; en[tid] = end_t[tid]; }
    __syncthreads();
    alpha[b][j] = st[j] + logits[(size_t)b * 16 + j];
    __syncthreads();
    for (int t = 1; t < S_; ++t) {
        float e = logits[((size_t)t * 32 + b) * 16 + j];
        float m = -1e30f;
#pragma unroll
        for (int i = 0; i < T_; ++i) m = fmaxf(m, alpha[b][i] + tr[i][j]);
        float s = 0.f;
#pragma unroll
        for (int i = 0; i < T_; ++i) s += __expf(alpha[b][i] + tr[i][j] - m);
        float nxt = m + __logf(s) + e;
        bool mk = labels[b * S_ + t] != 9;
        nxtb[b][j] = mk ? nxt : alpha[b][j];
        __syncthreads();
        alpha[b][j] = nxtb[b][j];
        __syncthreads();
    }
    if (j == 0) {
        float m = -1e30f;
        for (int i = 0; i < T_; ++i) m = fmaxf(m, alpha[b][i] + en[i]);
        float s = 0.f;
        for (int i = 0; i < T_; ++i) s += __expf(alpha[b][i] + en[i] - m);
        float denom = m + __logf(s);
        int lab0 = labels[b * S_];
        int tag0 = lab0 < 8 ? lab0 : 8;
        float num = st[tag0] + logits[(size_t)b * 16 + tag0];
        int prev = tag0, last = tag0;
        for (int t = 1; t < S_; ++t) {
            int lab = labels[b * S_ + t];
            int tg = lab < 8 ? lab : 8;
            float mf = (lab != 9) ? 1.0f : 0.0f;
            float e = logits[((size_t)t * 32 + b) * 16 + tg];
            num += (tr[prev][tg] + e) * mf;
            if (lab != 9) last = tg;
            prev = tg;
        }
        num += en[last];
        llh[b] = num - denom;
    }
    __syncthreads();
    if (tid == 0) {
        float s = 0.f;
        for (int bb = 0; bb < B_; ++bb) s += llh[bb];
        out_loss[0] = -(s / (float)B_);
    }
}

// ---- Viterbi decode body (one 320-thread block); decoded written as float
__device__ void viterbi_body(const float* __restrict__ logits,
                             const float* __restrict__ start_t,
                             const float* __restrict__ end_t,
                             const float* __restrict__ trans,
                             int* __restrict__ hist,
                             float* __restrict__ dec_out) {
    __shared__ float sc[B_][T_], nb[B_][T_];
    __shared__ float tr2[T_][T_], st2[T_], en2[T_];
    int tid = threadIdx.x;
    int b = tid / T_, j = tid % T_;
    if (tid < T_ * T_) tr2[tid / T_][tid % T_] = trans[tid];
    if (tid < T_) { st2[tid] = start_t[tid]; en2[tid] = end_t[tid]; }
    __syncthreads();
    sc[b][j] = st2[j] + logits[(size_t)b * 16 + j];
    __syncthreads();
    for (int t = 1; t < S_; ++t) {
        float e = logits[((size_t)t * 32 + b) * 16 + j];
        float best = -1e30f; int arg = 0;
#pragma unroll
        for (int i = 0; i < T_; ++i) {
            float v = sc[b][i] + tr2[i][j];
            if (v > best) { best = v; arg = i; }
        }
        nb[b][j] = best + e;
        hist[((size_t)(t - 1) * 32 + b) * T_ + j] = arg;
        __syncthreads();
        sc[b][j] = nb[b][j];
        __syncthreads();
    }
    if (j == 0) {
        float best = -1e30f; int last = 0;
        for (int i = 0; i < T_; ++i) {
            float v = sc[b][i] + en2[i];
            if (v > best) { best = v; last = i; }
        }
        dec_out[(size_t)b * S_ + (S_ - 1)] = (float)last;
        int tag = last;
        for (int t = S_ - 2; t >= 0; --t) {
            tag = hist[((size_t)t * 32 + b) * T_ + tag];
            dec_out[(size_t)b * S_ + t] = (float)tag;
        }
    }
}

// ---- fused: block 0 computes CRF loss, block 1 runs Viterbi (independent)
__global__ void __launch_bounds__(320) crf_viterbi(const float* __restrict__ logits,
                                                   const int* __restrict__ labels,
                                                   const float* __restrict__ start_t,
                                                   const float* __restrict__ end_t,
                                                   const float* __restrict__ trans,
                                                   int* __restrict__ hist,
                                                   float* __restrict__ out_loss,
                                                   float* __restrict__ dec_out) {
    if (blockIdx.x == 0)
        crf_loss_body(logits, labels, start_t, end_t, trans, out_loss);
    else
        viterbi_body(logits, start_t, end_t, trans, hist, dec_out);
}

extern "C" void kernel_launch(void* const* d_in, const int* in_sizes, int n_in,
                              void* d_out, int out_size, void* d_ws, size_t ws_size,
                              hipStream_t stream) {
    const int*   sents   = (const int*)d_in[0];
    const int*   lengths = (const int*)d_in[1];
    const int*   labels  = (const int*)d_in[2];
    const float* emb     = (const float*)d_in[3];
    const float* Wih_f   = (const float*)d_in[4];
    const float* Whh_f   = (const float*)d_in[5];
    const float* bih_f   = (const float*)d_in[6];
    const float* bhh_f   = (const float*)d_in[7];
    const float* Wih_b   = (const float*)d_in[8];
    const float* Whh_b   = (const float*)d_in[9];
    const float* bih_b   = (const float*)d_in[10];
    const float* bhh_b   = (const float*)d_in[11];
    const float* Wl      = (const float*)d_in[12];
    const float* bl      = (const float*)d_in[13];
    const float* start_t = (const float*)d_in[14];
    const float* end_t   = (const float*)d_in[15];
    const float* trans   = (const float*)d_in[16];

    char* ws = (char*)d_ws;
    bf16*  Apack  = (bf16*)(ws + OFF_APACK);
    bf16*  WihTf  = (bf16*)(ws + OFF_WIHF);
    bf16*  WhhTf  = (bf16*)(ws + OFF_WHHF);
    bf16*  WihTb  = (bf16*)(ws + OFF_WIHB);
    bf16*  WhhTb  = (bf16*)(ws + OFF_WHHB);
    bf16*  WlT    = (bf16*)(ws + OFF_WLT);
    bf16*  Gf     = (bf16*)(ws + OFF_GF);
    bf16*  Gb     = (bf16*)(ws + OFF_GB);
    float* Hout   = (float*)(ws + OFF_HOUT);
    bf16*  Hpack  = (bf16*)(ws + OFF_HPACK);
    float* logits = (float*)(ws + OFF_LOGITS);
    int*   hist   = (int*)(ws + OFF_HIST);
    float* loss_out = (float*)d_out;
    float* dec_out  = (float*)d_out + 1;

    // pack all weights into WMMA B fragments (W^T)
    pack_weightT<<<64, 256, 0, stream>>>(Wih_f, WihTf, 256, 1024, 1024);
    pack_weightT<<<64, 256, 0, stream>>>(Whh_f, WhhTf, 256, 1024, 1024);
    pack_weightT<<<64, 256, 0, stream>>>(Wih_b, WihTb, 256, 1024, 1024);
    pack_weightT<<<64, 256, 0, stream>>>(Whh_b, WhhTb, 256, 1024, 1024);
    pack_weightT<<<2, 256, 0, stream>>>(Wl, WlT, 512, 10, 16);

    // embedding gather -> A fragments
    pack_embed<<<1024, 256, 0, stream>>>(sents, emb, Apack);

    // input projections, both directions in one launch (grid.y = dir)
    gemm_proj2<<<dim3(4096, 2), 128, 0, stream>>>(Apack, WihTf, WihTb,
                                                  bih_f, bhh_f, bih_b, bhh_b, Gf, Gb);

    // both LSTM directions concurrently: 2 blocks x 32 wave32, 208KB LDS each
    size_t lstm_smem = 131072 + 32768 + 32768 + 16384;
    lstm_both<<<2, 1024, lstm_smem, stream>>>(Gf, Gb, WhhTf, WhhTb, lengths, Hout);

    // logits = H @ Wl^T + bl
    pack_hout<<<4096, 256, 0, stream>>>(Hout, Hpack);
    gemm_logits<<<256, 128, 0, stream>>>(Hpack, WlT, bl, logits);

    // CRF loss + Viterbi decode, concurrent in one launch
    crf_viterbi<<<2, 320, 0, stream>>>(logits, labels, start_t, end_t, trans,
                                       hist, loss_out, dec_out);
}
